// ToMeSequenceMerger_22780506538714
// MI455X (gfx1250) — matrix operand order repaired and checked by reading, not compiled, not run on hardware
//
#include <hip/hip_runtime.h>
#include <hip/hip_bf16.h>
#include <math.h>

#define L_SEQ 2048
#define DMODEL 768
#define NM_MAX (L_SEQ - 2)          /* 2046 interior tokens max */
#define NE_MAX ((NM_MAX + 1) / 2)   /* 1023 even rows max */

typedef __attribute__((ext_vector_type(16))) __bf16 v16bf;
typedef __attribute__((ext_vector_type(8)))  float  v8f;

/* ---------------- 1. mask-compact gather ---------------- */
__global__ void k_gather(const float* __restrict__ hid, const float* __restrict__ lcf,
                         const int* __restrict__ mask,
                         float* __restrict__ x, float* __restrict__ lf,
                         int* __restrict__ nbuf, int* __restrict__ vtmp) {
  int b = blockIdx.x;
  __shared__ int cnt_s;
  const int* mb = mask + (size_t)b * L_SEQ;
  int* vb = vtmp + (size_t)b * L_SEQ;
  if (threadIdx.x == 0) {
    int c = 0;
    for (int i = 0; i < L_SEQ; ++i)
      if (mb[i] > 0) { vb[c] = i; ++c; }
    nbuf[b] = c; cnt_s = c;
  }
  __syncthreads();
  int c = cnt_s;
  const float* hb = hid + (size_t)b * L_SEQ * DMODEL;
  float* xb = x + (size_t)b * L_SEQ * DMODEL;
  for (long idx = threadIdx.x; idx < (long)c * DMODEL; idx += blockDim.x) {
    int r = (int)(idx / DMODEL), d = (int)(idx % DMODEL);
    xb[(size_t)r * DMODEL + d] = hb[(size_t)vb[r] * DMODEL + d];
  }
  for (int r = threadIdx.x; r < c; r += blockDim.x)
    lf[(size_t)b * L_SEQ + r] = lcf[(size_t)b * L_SEQ + vb[r]];
}

/* ---------------- 2. normalize interior rows -> bf16 ---------------- */
__global__ void k_normalize(const float* __restrict__ x, unsigned short* __restrict__ m,
                            const int* __restrict__ nbuf) {
  int b = blockIdx.y, r = blockIdx.x;
  int n = nbuf[b]; int nm = n - 2;
  if (nm <= 0 || r >= nm) return;
  const float* row = x + ((size_t)b * L_SEQ + r + 1) * DMODEL;  /* PROT_L = 1 */
  __shared__ float red[256];
  float s = 0.f;
  for (int d = threadIdx.x; d < DMODEL; d += blockDim.x) { float v = row[d]; s += v * v; }
  red[threadIdx.x] = s; __syncthreads();
  for (int o = 128; o > 0; o >>= 1) {
    if ((int)threadIdx.x < o) red[threadIdx.x] += red[threadIdx.x + o];
    __syncthreads();
  }
  float inv = 1.0f / (sqrtf(red[0]) + 1e-6f);
  unsigned short* mrow = m + ((size_t)b * NM_MAX + r) * DMODEL;
  for (int d = threadIdx.x; d < DMODEL; d += blockDim.x) {
    unsigned int u = __float_as_uint(row[d] * inv);
    mrow[d] = (unsigned short)((u + 0x7FFFu + ((u >> 16) & 1u)) >> 16); /* RNE bf16 */
  }
}

/* ---------------- 3. WMMA cosine-score GEMM fused with row argmax ----------------
 * Each wave owns one 16-row tile of even tokens and sweeps all odd-token column
 * tiles 4 at a time: one A fragment per K-chunk feeds 4 independent v_wmma ops
 * (4 accumulator chains -> loads overlap matrix ops instead of draining loadcnt).
 */
__global__ void k_scores(const unsigned short* __restrict__ m, const int* __restrict__ nbuf,
                         int* __restrict__ assign) {
  int b = blockIdx.y;
  int wid = threadIdx.x >> 5;
  int lane = threadIdx.x & 31;
  int tile = blockIdx.x * 8 + wid;            /* wave-uniform row tile */
  int n = nbuf[b]; int nm = n - 2;
  if (nm < 2) return;
  int ne = (nm + 1) >> 1, no = nm >> 1;
  if (no <= 0) return;
  int M0 = tile * 16;
  if (M0 >= ne) return;

  const unsigned short* mb = m + (size_t)b * NM_MAX * DMODEL;
  int Ml = lane & 15;
  int hi = (lane >= 16) ? 1 : 0;
  int er = M0 + Ml; if (er > ne - 1) er = ne - 1;          /* clamp (results masked) */
  const unsigned short* arow = mb + (size_t)(2 * er) * DMODEL;   /* even token row */

  float bestv[8]; int bestj[8];
#pragma unroll
  for (int r = 0; r < 8; ++r) { bestv[r] = -3.4e38f; bestj[r] = 0; }

  int ntile = (no + 15) >> 4;
  for (int ctb = 0; ctb < ntile; ctb += 4) {
    int jj4[4];
    const unsigned short* brow[4];
    v8f c0 = {0.f,0.f,0.f,0.f,0.f,0.f,0.f,0.f};
    v8f c1 = c0, c2 = c0, c3 = c0;
#pragma unroll
    for (int t = 0; t < 4; ++t) {
      int ct = ctb + t; if (ct > ntile - 1) ct = ntile - 1;   /* clamp: dup tiles are harmless */
      int j = ct * 16 + (lane & 15);
      jj4[t] = j;
      int jc = (j > no - 1) ? (no - 1) : j;
      brow[t] = mb + (size_t)(2 * jc + 1) * DMODEL;           /* odd token row */
    }
    for (int kb = 0; kb < DMODEL; kb += 32) {
      v16bf a, b0, b1, b2, b3;
#pragma unroll
      for (int h = 0; h < 16; ++h) {
        int rr = h >> 1, p = h & 1;
        /* A 16x32 bf16 layout: M=lane%16; K in {0..7,16..23} (lo lanes) / +8 (hi lanes) */
        int ka = kb + (rr < 4 ? 2 * rr : 16 + 2 * (rr - 4)) + (hi ? 8 : 0) + p;
        union { unsigned short u; __bf16 f; } ca; ca.u = arow[ka]; a[h] = ca.f;
        /* B 32x16 bf16 layout: N=lane%16; K in {0..7,16..23} (lo lanes) / +8 (hi lanes) */
        int kc = kb + (h < 8 ? h : h + 8) + (hi ? 8 : 0);
        union { unsigned short u; __bf16 f; } u0; u0.u = brow[0][kc]; b0[h] = u0.f;
        union { unsigned short u; __bf16 f; } u1; u1.u = brow[1][kc]; b1[h] = u1.f;
        union { unsigned short u; __bf16 f; } u2; u2.u = brow[2][kc]; b2[h] = u2.f;
        union { unsigned short u; __bf16 f; } u3; u3.u = brow[3][kc]; b3[h] = u3.f;
      }
      c0 = __builtin_amdgcn_wmma_f32_16x16x32_bf16(false, a, false, b0, (short)0, c0, false, false);
      c1 = __builtin_amdgcn_wmma_f32_16x16x32_bf16(false, a, false, b1, (short)0, c1, false, false);
      c2 = __builtin_amdgcn_wmma_f32_16x16x32_bf16(false, a, false, b2, (short)0, c2, false, false);
      c3 = __builtin_amdgcn_wmma_f32_16x16x32_bf16(false, a, false, b3, (short)0, c3, false, false);
    }
    /* C layout: lane holds N=lane%16; VGPR r -> M = r + 8*hi. Running argmax.
     * Tiles processed in ascending j; strict > keeps argmax-first semantics. */
#pragma unroll
    for (int t = 0; t < 4; ++t) {
      int j = jj4[t];
      const v8f& c = (t == 0) ? c0 : (t == 1) ? c1 : (t == 2) ? c2 : c3;
#pragma unroll
      for (int r = 0; r < 8; ++r) {
        float v = c[r];
        if (j < no && v > bestv[r]) { bestv[r] = v; bestj[r] = j; }
      }
    }
  }
  /* reduce across the 16 lanes sharing the same M rows (halves stay separate) */
#pragma unroll
  for (int r = 0; r < 8; ++r) {
    float v = bestv[r]; int jj = bestj[r];
    for (int ofs = 1; ofs < 16; ofs <<= 1) {
      float ov = __shfl_xor(v, ofs, 32);
      int   oj = __shfl_xor(jj, ofs, 32);
      if (ov > v || (ov == v && oj < jj)) { v = ov; jj = oj; }  /* argmax-first ties */
    }
    int row = M0 + r + (hi ? 8 : 0);
    if ((lane & 15) == 0 && row < ne)
      assign[(size_t)b * NE_MAX + row] = jj;
  }
}

/* ---------------- 4. greedy dedup + merge + compaction (ping-pong) ---------------- */
__global__ void k_merge(const float* __restrict__ xsrc, float* __restrict__ xdst,
                        const float* __restrict__ lfsrc, float* __restrict__ lfdst,
                        int* __restrict__ nbuf, const int* __restrict__ assign) {
  int b = blockIdx.x;
  int n = nbuf[b];
  int nm = n - 2;
  __shared__ short partner[L_SEQ];
  __shared__ short newidx[L_SEQ];
  __shared__ unsigned char keepf[L_SEQ];
  __shared__ unsigned char claimed[NE_MAX + 1];
  __shared__ int as_s[NE_MAX];

  for (int i = threadIdx.x; i < L_SEQ; i += blockDim.x) { partner[i] = -1; keepf[i] = 1; }
  for (int i = threadIdx.x; i <= NE_MAX; i += blockDim.x) claimed[i] = 0;
  int ne = (nm >= 2) ? ((nm + 1) >> 1) : 0;
  int no = (nm >= 2) ? (nm >> 1) : 0;
  for (int i = threadIdx.x; i < ne; i += blockDim.x)
    as_s[i] = assign[(size_t)b * NE_MAX + i];
  __syncthreads();

  if (threadIdx.x == 0) {
    if (ne > 0 && no > 0) {
      for (int r = 0; r < ne; ++r) {          /* greedy: first row wins each column */
        int c = as_s[r];
        if (c >= 0 && c < no && !claimed[c]) {
          claimed[c] = 1;
          int src = 1 + 2 * r, dst = 2 + 2 * c;
          partner[src] = (short)dst;
          keepf[dst] = 0;
        }
      }
    }
    int np = 0;
    for (int r = 0; r < n; ++r) { if (keepf[r]) newidx[r] = (short)np++; else newidx[r] = -1; }
    nbuf[b] = np;
  }
  __syncthreads();

  const float* xs = xsrc + (size_t)b * L_SEQ * DMODEL;
  float* xd = xdst + (size_t)b * L_SEQ * DMODEL;
  for (long idx = threadIdx.x; idx < (long)n * DMODEL; idx += blockDim.x) {
    int r = (int)(idx / DMODEL), d = (int)(idx % DMODEL);
    if (!keepf[r]) continue;
    int dst = partner[r];
    float v = xs[(size_t)r * DMODEL + d];
    if (dst >= 0) v = 0.5f * (v + xs[(size_t)dst * DMODEL + d]);
    xd[(size_t)newidx[r] * DMODEL + d] = v;
  }
  const float* ls = lfsrc + (size_t)b * L_SEQ;
  float* ld = lfdst + (size_t)b * L_SEQ;
  for (int r = threadIdx.x; r < n; r += blockDim.x) {
    if (!keepf[r]) continue;
    int dst = partner[r];
    float lv = ls[r];
    if (dst >= 0) { float l2 = ls[dst]; lv = lv > l2 ? lv : l2; }
    ld[newidx[r]] = lv;
  }
}

/* ---------------- 5. linear resize back to L (align_corners=False) ---------------- */
__global__ void k_resize(const float* __restrict__ x, const float* __restrict__ lf,
                         const int* __restrict__ nbuf,
                         const float* __restrict__ hid, const float* __restrict__ lcf,
                         float* __restrict__ outh, float* __restrict__ outl) {
  int b = blockIdx.y, t = blockIdx.x;
  int n = nbuf[b];
  float* oh = outh + ((size_t)b * L_SEQ + t) * DMODEL;
  if (n <= 0) {  /* empty mask: pass through originals (reference behavior) */
    const float* hr = hid + ((size_t)b * L_SEQ + t) * DMODEL;
    for (int d = threadIdx.x; d < DMODEL; d += blockDim.x) oh[d] = hr[d];
    if (threadIdx.x == 0) outl[(size_t)b * L_SEQ + t] = lcf[(size_t)b * L_SEQ + t];
    return;
  }
  const float* xb = x + (size_t)b * L_SEQ * DMODEL;
  const float* lfb = lf + (size_t)b * L_SEQ;
  if (n == L_SEQ) {
    for (int d = threadIdx.x; d < DMODEL; d += blockDim.x) oh[d] = xb[(size_t)t * DMODEL + d];
    if (threadIdx.x == 0) outl[(size_t)b * L_SEQ + t] = lfb[t];
    return;
  }
  float coord = ((float)t + 0.5f) * ((float)n / (float)L_SEQ) - 0.5f;
  if (coord < 0.f) coord = 0.f;
  float mx = (float)(n - 1); if (coord > mx) coord = mx;
  int i0 = (int)floorf(coord);
  int i1 = i0 + 1; if (i1 > n - 1) i1 = n - 1;
  float w = coord - (float)i0;
  const float* r0 = xb + (size_t)i0 * DMODEL;
  const float* r1 = xb + (size_t)i1 * DMODEL;
  for (int d = threadIdx.x; d < DMODEL; d += blockDim.x)
    oh[d] = r0[d] * (1.0f - w) + r1[d] * w;
  if (threadIdx.x == 0)
    outl[(size_t)b * L_SEQ + t] = lfb[i0] * (1.0f - w) + lfb[i1] * w;
}

/* ---------------- launch ---------------- */
extern "C" void kernel_launch(void* const* d_in, const int* in_sizes, int n_in,
                              void* d_out, int out_size, void* d_ws, size_t ws_size,
                              hipStream_t stream) {
  const float* hid = (const float*)d_in[0];
  const float* lcf = (const float*)d_in[1];
  const int* mask = (const int*)d_in[2];
  int B = in_sizes[1] / L_SEQ;   /* in_sizes[1] = B*L */

  /* workspace partition (~151 MB for B=16) */
  char* w = (char*)d_ws;
  float* xA = (float*)w;             w += (size_t)B * L_SEQ * DMODEL * sizeof(float);
  float* lfA = (float*)w;            w += (size_t)B * L_SEQ * sizeof(float);
  float* lfB = (float*)w;            w += (size_t)B * L_SEQ * sizeof(float);
  unsigned short* m = (unsigned short*)w; w += (size_t)B * NM_MAX * DMODEL * sizeof(unsigned short);
  int* nbuf = (int*)w;               w += (size_t)B * sizeof(int) * 4; /* keep 16B align */
  int* assign = (int*)w;             w += (size_t)B * NE_MAX * sizeof(int);
  int* vtmp = (int*)w;               w += (size_t)B * L_SEQ * sizeof(int);

  float* outh = (float*)d_out;
  float* outl = outh + (size_t)B * L_SEQ * DMODEL;
  float* xB = outh;  /* d_out hidden region doubles as ping buffer; fully rewritten by k_resize */

  k_gather<<<dim3(B), 256, 0, stream>>>(hid, lcf, mask, xA, lfA, nbuf, vtmp);

  /* step 1: xA -> xB */
  k_normalize<<<dim3(NM_MAX, B), 256, 0, stream>>>(xA, m, nbuf);
  k_scores  <<<dim3(8, B),       256, 0, stream>>>(m, nbuf, assign);
  k_merge   <<<dim3(B),          256, 0, stream>>>(xA, xB, lfA, lfB, nbuf, assign);

  /* step 2: xB -> xA */
  k_normalize<<<dim3(NM_MAX, B), 256, 0, stream>>>(xB, m, nbuf);
  k_scores  <<<dim3(8, B),       256, 0, stream>>>(m, nbuf, assign);
  k_merge   <<<dim3(B),          256, 0, stream>>>(xB, xA, lfB, lfA, nbuf, assign);

  k_resize<<<dim3(L_SEQ, B), 256, 0, stream>>>(xA, lfA, nbuf, hid, lcf, outh, outl);
}
